// POFHPConv_34179349741573
// MI455X (gfx1250) — matrix operand order repaired and checked by reading, not compiled, not run on hardware
//
#include <hip/hip_runtime.h>
#include <math.h>

// ---------------- problem constants ----------------
#define U_N   20000
#define M_N   512
#define NS_N  4096
#define S_N   64
#define D_N   128
#define NN_N  20512          // U + M
#define EPOF  200000
#define EGAT  400000

typedef __bf16 bf16_t;
typedef __attribute__((ext_vector_type(16))) __bf16        bf16x16;
typedef __attribute__((ext_vector_type(8)))  float         f32x8;
typedef __attribute__((ext_vector_type(8)))  unsigned int  u32x8;

#define DEV static __device__ __forceinline__

// ---------------- small helpers ----------------
DEV bf16_t f2b(float f) {
  unsigned u = __builtin_bit_cast(unsigned, f);
  unsigned r = u + 0x7FFFu + ((u >> 16) & 1u);   // round-to-nearest-even
  unsigned short h = (unsigned short)(r >> 16);
  return __builtin_bit_cast(bf16_t, h);
}
DEV float b2f(bf16_t b) {
  unsigned short h = __builtin_bit_cast(unsigned short, b);
  unsigned u = ((unsigned)h) << 16;
  return __builtin_bit_cast(float, u);
}
DEV float sigm(float x)            { return 1.f / (1.f + __expf(-x)); }
DEV float eluf(float x)            { return x > 0.f ? x : __expf(x) - 1.f; }
DEV float lrelu(float x, float sl) { return x > 0.f ? x : sl * x; }
DEV float softplusf(float x)       { return (x > 20.f) ? x : log1pf(__expf(x)); }

DEV void atomicMaxF(float* addr, float val) {
  unsigned* ua = (unsigned*)addr;
  unsigned cur = *ua;
  while (__builtin_bit_cast(float, cur) < val) {
    unsigned prev = atomicCAS(ua, cur, __builtin_bit_cast(unsigned, val));
    if (prev == cur) break;
    cur = prev;
  }
}

// ---------------- WMMA fragment loaders (CDNA5 ISA 7.12.2 layouts) ----------------
// A (16x32 bf16, row-major source, row = lane&15):
//   half=0: V0..3 <- K = k0+0..7 ; V4..7 <- K = k0+16..23
//   half=1: V0..3 <- K = k0+8..15; V4..7 <- K = k0+24..31
DEV bf16x16 frag_a(const bf16_t* base, int ld, int k0, int lane) {
  int r = lane & 15, half = (lane >> 4) & 1;
  const unsigned int* p0 = (const unsigned int*)(base + r * ld + k0 + half * 8);
  const unsigned int* p1 = (const unsigned int*)(base + r * ld + k0 + 16 + half * 8);
  u32x8 u;
  u[0] = p0[0]; u[1] = p0[1]; u[2] = p0[2]; u[3] = p0[3];
  u[4] = p1[0]; u[5] = p1[1]; u[6] = p1[2]; u[7] = p1[3];
  return __builtin_bit_cast(bf16x16, u);
}
// B (32x16 bf16) built from weight W stored [N,K] row-major: B[k][n] = W[n][k].
//   lane = column n (lane&15); half=0 -> K k0+0..7, k0+16..23; half=1 -> K k0+8..15, k0+24..31 (same as A)
DEV bf16x16 frag_b(const bf16_t* base, int ld, int k0, int lane) {
  int r = lane & 15, half = (lane >> 4) & 1;
  const unsigned int* p = (const unsigned int*)(base + r * ld + k0 + half * 8);
  u32x8 u;
  u[0] = p[0]; u[1] = p[1]; u[2] = p[2]; u[3] = p[3];
  u[4] = p[8]; u[5] = p[9]; u[6] = p[10]; u[7] = p[11];
  return __builtin_bit_cast(bf16x16, u);
}
DEV f32x8 wmma_bf16(bf16x16 a, bf16x16 b, f32x8 c) {
  return __builtin_amdgcn_wmma_f32_16x16x32_bf16(false, a, false, b, (short)0, c,
                                                 false, false);
}

// ---------------- utility kernels ----------------
__global__ void fill_f32_kernel(float* p, float v, int n) {
  int i = blockIdx.x * blockDim.x + threadIdx.x;
  if (i < n) p[i] = v;
}
__global__ void cvt_bf16_kernel(const float* s, bf16_t* d, int n) {
  int i = blockIdx.x * blockDim.x + threadIdx.x;
  if (i < n) d[i] = f2b(s[i]);
}

// ---------------- generic GEMM:  C[M,N] = A[M,128] @ W[N,128]^T (+bias) ----------------
__global__ __launch_bounds__(128) void gemm_xwT_kernel(
    const bf16_t* __restrict__ A, const bf16_t* __restrict__ W,
    const float* __restrict__ bias, float* __restrict__ C, bf16_t* __restrict__ Cb,
    int Mrows, int Ncols) {
  const int K = 128;
  int wave = threadIdx.x >> 5;
  int lane = threadIdx.x & 31;
  int tilesN = Ncols >> 4;
  int tile = blockIdx.x * 4 + wave;
  int total = (Mrows >> 4) * tilesN;
  if (tile >= total) return;
  int tm = tile / tilesN, tn = tile % tilesN;
  const bf16_t* Ap = A + (size_t)tm * 16 * K;
  const bf16_t* Wp = W + (size_t)tn * 16 * K;
  __builtin_prefetch(Ap, 0, 0);
  __builtin_prefetch(Wp, 0, 0);
  f32x8 acc;
#pragma unroll
  for (int v = 0; v < 8; ++v) acc[v] = 0.f;
#pragma unroll
  for (int k0 = 0; k0 < K; k0 += 32) {
    bf16x16 a = frag_a(Ap, K, k0, lane);
    bf16x16 b = frag_b(Wp, K, k0, lane);
    acc = wmma_bf16(a, b, acc);
  }
  int r = lane & 15, half = (lane >> 4) & 1;
  float bv = bias ? bias[tn * 16 + r] : 0.f;
#pragma unroll
  for (int v = 0; v < 8; ++v) {
    int row = tm * 16 + v + 8 * half;
    float val = acc[v] + bv;
    size_t idx = (size_t)row * Ncols + tn * 16 + r;
    C[idx] = val;
    if (Cb) Cb[idx] = f2b(val);
  }
}

// ---------------- fused GRU (+ w_his / w_intra dots) ----------------
// 16 sequences per workgroup; H tile in LDS; per-step: GI = X@Wih^T, GH = H@Whh^T via WMMA.
__global__ __launch_bounds__(256) void gru_kernel(
    const bf16_t* __restrict__ ux_bf, const int* __restrict__ user_series,
    const bf16_t* __restrict__ wih, const bf16_t* __restrict__ whh,
    const float* __restrict__ b_ih, const float* __restrict__ b_hh,
    const float* __restrict__ w_his, const float* __restrict__ w_intra,
    float* __restrict__ wall_out, float* __restrict__ f_out) {
  __shared__ bf16_t Xl[16 * 128];
  __shared__ bf16_t Hl[16 * 128];
  __shared__ float  GI[16 * 384];
  __shared__ float  GH[16 * 384];
  int tid = threadIdx.x;
  int wave = tid >> 5, lane = tid & 31;
  int seqbase = blockIdx.x * 16;
  for (int i = tid; i < 16 * 128; i += 256) Hl[i] = f2b(0.f);
  __syncthreads();
  int row = tid >> 4;           // 0..15
  int c8  = (tid & 15) * 8;     // 8-col slice for gather / elementwise
  for (int t = 0; t < S_N; ++t) {
    { // gather x_t rows (bf16, 16B per thread)
      int idx = user_series[(size_t)(seqbase + row) * S_N + t];
      const unsigned int* src = (const unsigned int*)(ux_bf + (size_t)idx * 128 + c8);
      unsigned int* dst = (unsigned int*)(Xl + row * 128 + c8);
      dst[0] = src[0]; dst[1] = src[1]; dst[2] = src[2]; dst[3] = src[3];
    }
    __syncthreads();
    // gate GEMMs: 24 col-tiles of 16, 3 per wave
#pragma unroll
    for (int tt = 0; tt < 3; ++tt) {
      int n0 = (wave * 3 + tt) * 16;
      f32x8 ai, ah;
#pragma unroll
      for (int v = 0; v < 8; ++v) { ai[v] = 0.f; ah[v] = 0.f; }
#pragma unroll
      for (int k0 = 0; k0 < 128; k0 += 32) {
        bf16x16 ax = frag_a(Xl, 128, k0, lane);
        bf16x16 ah_ = frag_a(Hl, 128, k0, lane);
        bf16x16 bi = frag_b(wih + (size_t)n0 * 128, 128, k0, lane);
        bf16x16 bh = frag_b(whh + (size_t)n0 * 128, 128, k0, lane);
        ai = wmma_bf16(ax, bi, ai);
        ah = wmma_bf16(ah_, bh, ah);
      }
      int r = lane & 15, half = (lane >> 4) & 1;
#pragma unroll
      for (int v = 0; v < 8; ++v) {
        GI[(v + 8 * half) * 384 + n0 + r] = ai[v];
        GH[(v + 8 * half) * 384 + n0 + r] = ah[v];
      }
    }
    __syncthreads();
    // elementwise gate math + fused dots with w_his / w_intra
    float wpart = 0.f, fpart = 0.f;
#pragma unroll
    for (int i = 0; i < 8; ++i) {
      int c = c8 + i;
      float ir = GI[row * 384 + c]       + b_ih[c];
      float iz = GI[row * 384 + 128 + c] + b_ih[128 + c];
      float in_ = GI[row * 384 + 256 + c] + b_ih[256 + c];
      float hr = GH[row * 384 + c]       + b_hh[c];
      float hz = GH[row * 384 + 128 + c] + b_hh[128 + c];
      float hn = GH[row * 384 + 256 + c] + b_hh[256 + c];
      float rg = sigm(ir + hr);
      float zg = sigm(iz + hz);
      float ng = tanhf(in_ + rg * hn);
      float hold = b2f(Hl[row * 128 + c]);
      float hnew = (1.f - zg) * ng + zg * hold;
      Hl[row * 128 + c] = f2b(hnew);
      wpart += hnew * w_his[c];
      fpart += hnew * w_intra[c];
    }
#pragma unroll
    for (int m = 8; m >= 1; m >>= 1) {
      wpart += __shfl_xor(wpart, m, 16);
      fpart += __shfl_xor(fpart, m, 16);
    }
    if ((tid & 15) == 0) {
      wall_out[(size_t)(seqbase + row) * S_N + t] = wpart;
      f_out[(size_t)(seqbase + row) * S_N + t]    = fpart;
    }
    __syncthreads();
  }
}

// ---------------- message dots + gi scalar ----------------
__global__ void mdots_kernel(const float* mx_f, const float* w_base, const float* w_inter,
                             const float* mft, float* mdot_base, float* mdot_inter,
                             float* gi_scalar) {
  __shared__ float sb[128], si[128];
  int m = blockIdx.x, d = threadIdx.x;
  float x = mx_f[(size_t)m * 128 + d];
  sb[d] = x * w_base[d];
  si[d] = x * w_inter[d];
  __syncthreads();
  for (int s = 64; s > 0; s >>= 1) {
    if (d < s) { sb[d] += sb[d + s]; si[d] += si[d + s]; }
    __syncthreads();
  }
  if (d == 0) {
    mdot_base[m] = sb[0];
    mdot_inter[m] = si[0];
    atomicAdd(gi_scalar, __expf(2.f * mft[m]) * si[0]);
  }
}

// ---------------- pofe: cif evaluated only at seq_last_idx ----------------
__global__ void pofe_kernel(const float* wall, const float* fband, const float* ts,
                            const float* slt, const int* sli, const int* seq_idx,
                            const float* mdot_base, const float* mdot_inter,
                            const float* gi_scalar, float* pofe) {
  int ns = blockIdx.x * blockDim.x + threadIdx.x;
  if (ns >= NS_N) return;
  int li = sli[ns];
  int m = seq_idx[ns];
  float tli = ts[(size_t)ns * S_N + li];
  float X1 = 0.f;
  for (int j = 0; j < li; ++j)
    X1 += __expf(-2.f * (tli - ts[(size_t)ns * S_N + j])) * wall[(size_t)ns * S_N + j];
  float X2 = 0.f;
  for (int q = 0; q < 8; ++q) {
    int ns2 = m * 8 + q;
    X2 += __expf(2.f * slt[ns2]) * fband[(size_t)ns2 * S_N + li];
  }
  X2 *= __expf(-2.f * slt[ns]);           // cascade size is always 8 (>1)
  float X3 = __expf(-2.f * tli) * gi_scalar[0] - mdot_inter[m];
  float c = sigm(X1) + sigm(X2) + sigm(X3) + sigm(mdot_base[m]);
  atomicAdd(&pofe[m], softplusf(c));
}

__global__ void pofe_norm_kernel(const float* pofe, float* pofen) {
  __shared__ float sh[512];
  int m = threadIdx.x;
  sh[m] = pofe[m];
  __syncthreads();
  for (int k = 256; k > 0; k >>= 1) {
    if (m < k) sh[m] += sh[m + k];
    __syncthreads();
  }
  pofen[m] = pofe[m] / sh[0];
}

// ---------------- attention-vector dots for pof / interest ----------------
__global__ void attvec_kernel(const float* px_f, const float* poc, const float* itt,
                              float* asrc, float* adst, float* bsrc, float* bdst, int n) {
  int i = blockIdx.x * blockDim.x + threadIdx.x;
  if (i >= n) return;
  float a = 0.f, b = 0.f, c = 0.f, e = 0.f;
  for (int d = 0; d < 128; ++d) {
    float x = px_f[(size_t)i * 128 + d];
    a += x * poc[d]; b += x * poc[128 + d];
    c += x * itt[d]; e += x * itt[128 + d];
  }
  asrc[i] = a; adst[i] = b; bsrc[i] = c; bdst[i] = e;
}

// ---------------- pof_learning edge passes (only src >= U feeds output) ----------------
__global__ void pof_edge_vals_kernel(const int* ei, const float* asrc, const float* adst,
                                     const float* rt, const float* t_o,
                                     float* ev, float* rmax) {
  int e = blockIdx.x * blockDim.x + threadIdx.x;
  if (e >= EPOF) return;
  int s = ei[e];
  if (s < U_N) { ev[e] = 0.f; return; }
  int d = ei[EPOF + e];
  float v = lrelu(asrc[s] + adst[d], 0.01f) * __expf(-2.f * (t_o[0] - rt[e]));
  float nv = -v;
  ev[e] = nv;
  atomicMaxF(&rmax[s - U_N], nv);
}
__global__ void pof_edge_exp_kernel(const int* ei, const float* ev, const float* rmax,
                                    float* ee, float* rsum) {
  int e = blockIdx.x * blockDim.x + threadIdx.x;
  if (e >= EPOF) return;
  int s = ei[e];
  if (s < U_N) return;
  float x = __expf(ev[e] - rmax[s - U_N]);
  ee[e] = x;
  atomicAdd(&rsum[s - U_N], x);
}
__global__ void pof_edge_acc_kernel(const int* ei, const float* ee, const float* rsum,
                                    const float* px_f, float* pm) {
  int e = blockIdx.x;
  int s = ei[e];
  if (s >= U_N) {
    int dd = ei[EPOF + e];
    float w = ee[e] / fmaxf(rsum[s - U_N], 1e-16f);
    int d = threadIdx.x;
    atomicAdd(&pm[(size_t)(s - U_N) * 128 + d], w * px_f[(size_t)dd * 128 + d]);
  }
}

// ---------------- user_interest (self loops folded into init) ----------------
__global__ void int_self_init_kernel(const float* bsrc, const float* bdst,
                                     float* selfv, float* imax) {
  int i = blockIdx.x * blockDim.x + threadIdx.x;
  if (i >= U_N) return;
  float nv = -lrelu(bsrc[i] + bdst[i], 0.01f);
  selfv[i] = nv;
  imax[i] = nv;
}
__global__ void int_edge_max_kernel(const int* ei, const float* bsrc, const float* bdst,
                                    float* imax) {
  int e = blockIdx.x * blockDim.x + threadIdx.x;
  if (e >= EPOF) return;
  int s = ei[e];
  if (s >= U_N) return;
  int d = ei[EPOF + e];
  atomicMaxF(&imax[s], -lrelu(bsrc[s] + bdst[d], 0.01f));
}
__global__ void int_sum_init_kernel(const float* selfv, const float* imax, float* isum) {
  int i = blockIdx.x * blockDim.x + threadIdx.x;
  if (i >= U_N) return;
  isum[i] = __expf(selfv[i] - imax[i]);
}
__global__ void int_edge_sum_kernel(const int* ei, const float* bsrc, const float* bdst,
                                    const float* imax, float* isum) {
  int e = blockIdx.x * blockDim.x + threadIdx.x;
  if (e >= EPOF) return;
  int s = ei[e];
  if (s >= U_N) return;
  int d = ei[EPOF + e];
  atomicAdd(&isum[s], __expf(-lrelu(bsrc[s] + bdst[d], 0.01f) - imax[s]));
}
__global__ void int_edge_acc_kernel(const int* ei, const float* bsrc, const float* bdst,
                                    const float* imax, const float* isum,
                                    const float* px_f, const float* pm,
                                    const float* pofen, float* uacc) {
  int e = blockIdx.x;
  int s = ei[e];
  if (s >= U_N) return;
  int dd = ei[EPOF + e];
  float w = __expf(-lrelu(bsrc[s] + bdst[dd], 0.01f) - imax[s]) / fmaxf(isum[s], 1e-16f);
  int d = threadIdx.x;
  float xv;
  if (dd < U_N) {
    xv = px_f[(size_t)dd * 128 + d];
  } else {
    int j = dd - U_N;
    xv = px_f[(size_t)dd * 128 + d] + pofen[j] * eluf(pm[(size_t)j * 128 + d]);
  }
  atomicAdd(&uacc[(size_t)s * 128 + d], w * xv);
}
__global__ void int_self_acc_kernel(const float* selfv, const float* imax,
                                    const float* isum, const float* px_f, float* uacc) {
  int i = blockIdx.x;
  float w = __expf(selfv[i] - imax[i]) / fmaxf(isum[i], 1e-16f);
  int d = threadIdx.x;
  atomicAdd(&uacc[(size_t)i * 128 + d], w * px_f[(size_t)i * 128 + d]);
}

// ---------------- GAT (shared by gat1 / gat2) ----------------
__global__ void gat_sd_kernel(const float* h, const float* as_, const float* ad_,
                              float* sarr, float* darr, int n) {
  int id = blockIdx.x * blockDim.x + threadIdx.x;
  if (id >= n * 4) return;
  int i = id >> 2, hh = id & 3;
  float sv = 0.f, dv = 0.f;
  for (int k = 0; k < 128; ++k) {
    float x = h[(size_t)i * 512 + hh * 128 + k];
    sv += x * as_[hh * 128 + k];
    dv += x * ad_[hh * 128 + k];
  }
  sarr[id] = sv;
  darr[id] = dv;
}
__global__ void gat_self_init_kernel(const float* sarr, const float* darr,
                                     float* dmax, int n) {
  int id = blockIdx.x * blockDim.x + threadIdx.x;
  if (id >= n * 4) return;
  dmax[id] = lrelu(sarr[id] + darr[id], 0.2f);
}
__global__ void gat_edge_max_kernel(const int* ei, int E, const float* sarr,
                                    const float* darr, float* dmax) {
  int e = blockIdx.x * blockDim.x + threadIdx.x;
  if (e >= E) return;
  int sr = ei[e], ds = ei[E + e];
  for (int hh = 0; hh < 4; ++hh)
    atomicMaxF(&dmax[ds * 4 + hh], lrelu(sarr[sr * 4 + hh] + darr[ds * 4 + hh], 0.2f));
}
__global__ void gat_sum_init_kernel(const float* sarr, const float* darr,
                                    const float* dmax, float* dsum, int n) {
  int id = blockIdx.x * blockDim.x + threadIdx.x;
  if (id >= n * 4) return;
  dsum[id] = __expf(lrelu(sarr[id] + darr[id], 0.2f) - dmax[id]);
}
__global__ void gat_edge_sum_kernel(const int* ei, int E, const float* sarr,
                                    const float* darr, const float* dmax, float* dsum) {
  int e = blockIdx.x * blockDim.x + threadIdx.x;
  if (e >= E) return;
  int sr = ei[e], ds = ei[E + e];
  for (int hh = 0; hh < 4; ++hh)
    atomicAdd(&dsum[ds * 4 + hh],
              __expf(lrelu(sarr[sr * 4 + hh] + darr[ds * 4 + hh], 0.2f) - dmax[ds * 4 + hh]));
}
__global__ void gat_edge_acc_kernel(const int* ei, int E, const float* sarr,
                                    const float* darr, const float* dmax,
                                    const float* dsum, const float* h, float* gacc) {
  int e = blockIdx.x;
  int sr = ei[e], ds = ei[E + e];
  float w[4];
  for (int hh = 0; hh < 4; ++hh)
    w[hh] = 0.25f *
            __expf(lrelu(sarr[sr * 4 + hh] + darr[ds * 4 + hh], 0.2f) - dmax[ds * 4 + hh]) /
            fmaxf(dsum[ds * 4 + hh], 1e-16f);
  int d = threadIdx.x;
  float a = 0.f;
  for (int hh = 0; hh < 4; ++hh) a += w[hh] * h[(size_t)sr * 512 + hh * 128 + d];
  atomicAdd(&gacc[(size_t)ds * 128 + d], a);
}
__global__ void gat_self_acc_kernel(const float* sarr, const float* darr,
                                    const float* dmax, const float* dsum,
                                    const float* h, float* gacc) {
  int i = blockIdx.x;
  float w[4];
  for (int hh = 0; hh < 4; ++hh)
    w[hh] = 0.25f *
            __expf(lrelu(sarr[i * 4 + hh] + darr[i * 4 + hh], 0.2f) - dmax[i * 4 + hh]) /
            fmaxf(dsum[i * 4 + hh], 1e-16f);
  int d = threadIdx.x;
  float a = 0.f;
  for (int hh = 0; hh < 4; ++hh) a += w[hh] * h[(size_t)i * 512 + hh * 128 + d];
  atomicAdd(&gacc[(size_t)i * 128 + d], a);
}

// ---------------- fused single-query attention (h_pr[:,0,:] only) ----------------
__global__ __launch_bounds__(256) void attn_kernel(
    const bf16_t* __restrict__ ux_bf, const float* __restrict__ mx_f,
    const int* __restrict__ user_series, const int* __restrict__ seq_idx,
    const float* __restrict__ ts, const float* __restrict__ mask,
    const bf16_t* __restrict__ KWb, const bf16_t* __restrict__ VWb,
    const float* __restrict__ QW, const float* __restrict__ Qb,
    const float* __restrict__ Kb, const float* __restrict__ Vb,
    const float* __restrict__ ftW, const float* __restrict__ ftb,
    float* __restrict__ longacc) {
  __shared__ bf16_t Hq[80 * 128];
  __shared__ bf16_t Kl[80 * 128];
  __shared__ bf16_t Vl[80 * 128];
  __shared__ float  Q0[128];
  __shared__ float  sc[65];
  __shared__ float  red[1];
  int ns = blockIdx.x;
  int m = seq_idx[ns];
  int tid = threadIdx.x;
  // build Hq (row 0 = x_mk_b; rows 1..64 = usf + time encoding; 65..79 = zero pad)
  for (int i = tid; i < 80 * 128; i += 256) {
    int rrow = i >> 7, d = i & 127;
    float v;
    if (rrow == 0) v = mx_f[(size_t)m * 128 + d];
    else if (rrow <= 64) {
      int t = rrow - 1;
      int idx = user_series[(size_t)ns * S_N + t];
      v = b2f(ux_bf[(size_t)idx * 128 + d]) + ts[(size_t)ns * S_N + t] * ftW[d] + ftb[d];
    } else v = 0.f;
    Hq[i] = f2b(v);
  }
  __syncthreads();
  int wave = tid >> 5, lane = tid & 31;
  // K and V tiles: 5 row-tiles x 8 col-tiles x 2 matrices = 80 tiles, 10 per wave
  for (int jt = wave; jt < 80; jt += 8) {
    int mat = jt / 40;
    int tile = jt % 40;
    int tm = tile >> 3, tn = tile & 7;
    const bf16_t* Wb = mat ? VWb : KWb;
    const float* bb = mat ? Vb : Kb;
    bf16_t* out = mat ? Vl : Kl;
    f32x8 acc;
#pragma unroll
    for (int v = 0; v < 8; ++v) acc[v] = 0.f;
#pragma unroll
    for (int k0 = 0; k0 < 128; k0 += 32) {
      bf16x16 a = frag_a(Hq + tm * 16 * 128, 128, k0, lane);
      bf16x16 b = frag_b(Wb + (size_t)tn * 16 * 128, 128, k0, lane);
      acc = wmma_bf16(a, b, acc);
    }
    int r = lane & 15, half = (lane >> 4) & 1;
#pragma unroll
    for (int v = 0; v < 8; ++v) {
      int row = tm * 16 + v + 8 * half;
      out[row * 128 + tn * 16 + r] = f2b(acc[v] + bb[tn * 16 + r]);
    }
  }
  if (tid < 128) {  // Q for row 0 only (128x128 dot)
    float a = 0.f;
    for (int k = 0; k < 128; ++k) a += mx_f[(size_t)m * 128 + k] * QW[tid * 128 + k];
    Q0[tid] = a + Qb[tid];
  }
  __syncthreads();
  if (tid < 65) {
    float s = 0.f;
    for (int d = 0; d < 128; ++d) s += Q0[d] * b2f(Kl[tid * 128 + d]);
    s *= 0.08838834764831845f;  // 1/sqrt(128)
    float mk = (tid == 0) ? 1.f : mask[(size_t)ns * S_N + tid - 1];
    sc[tid] = (mk == 1.f) ? s : -INFINITY;
  }
  __syncthreads();
  if (tid == 0) {
    float mxv = -INFINITY;
    for (int t = 0; t < 65; ++t) mxv = fmaxf(mxv, sc[t]);
    float sm = 0.f;
    for (int t = 0; t < 65; ++t) { float e = __expf(sc[t] - mxv); sc[t] = e; sm += e; }
    red[0] = sm;
  }
  __syncthreads();
  if (tid < 128) {
    float inv = 1.f / red[0];
    float acc = 0.f;
    for (int t = 0; t < 65; ++t) acc += sc[t] * b2f(Vl[t * 128 + tid]);
    atomicAdd(&longacc[(size_t)m * 128 + tid], acc * inv * 0.125f);  // counts == 8
  }
}

// ---------------- outputs ----------------
__global__ void out_msg_kernel(const float* pm, const float* longacc, float* out) {
  int i = blockIdx.x * blockDim.x + threadIdx.x;
  if (i >= M_N * 128) return;
  out[i] = eluf(pm[i]) + eluf(longacc[i]);
}
__global__ void out_user_kernel(const float* gacc1, const float* g1b, const float* uacc,
                                const float* gacc2, const float* g2b, float* out) {
  int id = blockIdx.x * blockDim.x + threadIdx.x;
  if (id >= U_N * 128) return;
  int i = id >> 7, d = id & 127;
  size_t o = (size_t)i * 384;
  out[o + d]       = eluf(gacc1[id] + g1b[d]);
  out[o + 128 + d] = eluf(uacc[id]);
  out[o + 256 + d] = eluf(gacc2[id] + g2b[d]);
}

// ================= host launcher =================
extern "C" void kernel_launch(void* const* d_in, const int* in_sizes, int n_in,
                              void* d_out, int out_size, void* d_ws, size_t ws_size,
                              hipStream_t stream) {
  (void)in_sizes; (void)n_in; (void)out_size; (void)ws_size;
  const float* user_x   = (const float*)d_in[0];
  const float* msg_feat = (const float*)d_in[1];
  const float* pof_x    = (const float*)d_in[2];
  const int*   u_series = (const int*)d_in[3];
  const int*   seq_idx  = (const int*)d_in[4];
  const float* ts       = (const float*)d_in[5];
  const int*   sli      = (const int*)d_in[6];
  const float* slt      = (const float*)d_in[7];
  const float* mask     = (const float*)d_in[8];
  const float* mft      = (const float*)d_in[9];
  const int*   pof_ei   = (const int*)d_in[10];
  const float* rt       = (const float*)d_in[11];
  const float* t_o      = (const float*)d_in[12];
  const int*   pr_ei    = (const int*)d_in[13];
  const int*   fr_ei    = (const int*)d_in[14];
  const float* proj_W   = (const float*)d_in[15];
  const float* proj_b   = (const float*)d_in[16];
  const float* w_ih     = (const float*)d_in[17];
  const float* w_hh     = (const float*)d_in[18];
  const float* b_ih     = (const float*)d_in[19];
  const float* b_hh     = (const float*)d_in[20];
  const float* w_his    = (const float*)d_in[21];
  const float* w_intra  = (const float*)d_in[22];
  const float* w_inter  = (const float*)d_in[23];
  const float* w_base   = (const float*)d_in[24];
  const float* poc_att  = (const float*)d_in[25];
  const float* int_att  = (const float*)d_in[26];
  const float* QW       = (const float*)d_in[27];
  const float* Qb       = (const float*)d_in[28];
  const float* KW       = (const float*)d_in[29];
  const float* Kb       = (const float*)d_in[30];
  const float* VW       = (const float*)d_in[31];
  const float* Vb       = (const float*)d_in[32];
  const float* ftW      = (const float*)d_in[33];
  const float* ftb      = (const float*)d_in[34];
  const float* g1W      = (const float*)d_in[35];
  const float* g1as     = (const float*)d_in[36];
  const float* g1ad     = (const float*)d_in[37];
  const float* g1b      = (const float*)d_in[38];
  const float* g2W      = (const float*)d_in[39];
  const float* g2as     = (const float*)d_in[40];
  const float* g2ad     = (const float*)d_in[41];
  const float* g2b      = (const float*)d_in[42];

  char* base = (char*)d_ws;
  size_t off = 0;
  auto alloc = [&](size_t nbytes) -> void* {
    void* p = base + off;
    off += (nbytes + 255) & ~(size_t)255;
    return p;
  };
  bf16_t* xa_bf  = (bf16_t*)alloc((size_t)U_N * 128 * 2);
  bf16_t* xm_bf  = (bf16_t*)alloc((size_t)M_N * 128 * 2);
  bf16_t* xp_bf  = (bf16_t*)alloc((size_t)NN_N * 128 * 2);
  bf16_t* wb_proj = (bf16_t*)alloc(128 * 128 * 2);
  bf16_t* wb_ih  = (bf16_t*)alloc(384 * 128 * 2);
  bf16_t* wb_hh  = (bf16_t*)alloc(384 * 128 * 2);
  bf16_t* wb_K   = (bf16_t*)alloc(128 * 128 * 2);
  bf16_t* wb_V   = (bf16_t*)alloc(128 * 128 * 2);
  bf16_t* wb_g1  = (bf16_t*)alloc(512 * 128 * 2);
  bf16_t* wb_g2  = (bf16_t*)alloc(512 * 128 * 2);
  float* ux_f    = (float*)alloc((size_t)U_N * 128 * 4);
  float* mx_f    = (float*)alloc((size_t)M_N * 128 * 4);
  float* px_f    = (float*)alloc((size_t)NN_N * 128 * 4);
  bf16_t* ux_bf  = (bf16_t*)alloc((size_t)U_N * 128 * 2);
  float* h1      = (float*)alloc((size_t)U_N * 512 * 4);
  float* h2      = (float*)alloc((size_t)U_N * 512 * 4);
  float* wall    = (float*)alloc((size_t)NS_N * S_N * 4);
  float* fband   = (float*)alloc((size_t)NS_N * S_N * 4);
  float* mdb     = (float*)alloc(M_N * 4);
  float* mdi     = (float*)alloc(M_N * 4);
  float* gi_s    = (float*)alloc(4);
  float* pofe    = (float*)alloc(M_N * 4);
  float* pofen   = (float*)alloc(M_N * 4);
  float* asrc    = (float*)alloc((size_t)NN_N * 4);
  float* adst    = (float*)alloc((size_t)NN_N * 4);
  float* bsrc    = (float*)alloc((size_t)NN_N * 4);
  float* bdst    = (float*)alloc((size_t)NN_N * 4);
  float* ev      = (float*)alloc((size_t)EPOF * 4);
  float* ee      = (float*)alloc((size_t)EPOF * 4);
  float* rmax    = (float*)alloc(M_N * 4);
  float* rsum    = (float*)alloc(M_N * 4);
  float* pm      = (float*)alloc((size_t)M_N * 128 * 4);
  float* selfv   = (float*)alloc((size_t)U_N * 4);
  float* imax    = (float*)alloc((size_t)U_N * 4);
  float* isum    = (float*)alloc((size_t)U_N * 4);
  float* uacc    = (float*)alloc((size_t)U_N * 128 * 4);
  float* s1      = (float*)alloc((size_t)U_N * 4 * 4);
  float* d1      = (float*)alloc((size_t)U_N * 4 * 4);
  float* s2      = (float*)alloc((size_t)U_N * 4 * 4);
  float* d2      = (float*)alloc((size_t)U_N * 4 * 4);
  float* dmax1   = (float*)alloc((size_t)U_N * 4 * 4);
  float* dsum1   = (float*)alloc((size_t)U_N * 4 * 4);
  float* dmax2   = (float*)alloc((size_t)U_N * 4 * 4);
  float* dsum2   = (float*)alloc((size_t)U_N * 4 * 4);
  float* gacc1   = (float*)alloc((size_t)U_N * 128 * 4);
  float* gacc2   = (float*)alloc((size_t)U_N * 128 * 4);
  float* longacc = (float*)alloc((size_t)M_N * 128 * 4);

  auto cvt = [&](const float* s, bf16_t* d, int n) {
    cvt_bf16_kernel<<<dim3((n + 255) / 256), dim3(256), 0, stream>>>(s, d, n);
  };
  auto fill = [&](float* p, float v, int n) {
    fill_f32_kernel<<<dim3((n + 255) / 256), dim3(256), 0, stream>>>(p, v, n);
  };
  auto gemm = [&](const bf16_t* A, const bf16_t* W, const float* bias, float* C,
                  bf16_t* Cb, int Mr, int Nc) {
    int tiles = (Mr / 16) * (Nc / 16);
    gemm_xwT_kernel<<<dim3((tiles + 3) / 4), dim3(128), 0, stream>>>(A, W, bias, C, Cb,
                                                                     Mr, Nc);
  };

  // 1. bf16 conversions
  cvt(user_x, xa_bf, U_N * 128);
  cvt(msg_feat, xm_bf, M_N * 128);
  cvt(pof_x, xp_bf, NN_N * 128);
  cvt(proj_W, wb_proj, 128 * 128);
  cvt(w_ih, wb_ih, 384 * 128);
  cvt(w_hh, wb_hh, 384 * 128);
  cvt(KW, wb_K, 128 * 128);
  cvt(VW, wb_V, 128 * 128);
  cvt(g1W, wb_g1, 512 * 128);
  cvt(g2W, wb_g2, 512 * 128);

  // 2. projections (WMMA), dual fp32+bf16 for ux
  gemm(xa_bf, wb_proj, proj_b, ux_f, ux_bf, U_N, 128);
  gemm(xm_bf, wb_proj, proj_b, mx_f, nullptr, M_N, 128);
  gemm(xp_bf, wb_proj, proj_b, px_f, nullptr, NN_N, 128);

  // 3. GAT feature transforms (WMMA)
  gemm(ux_bf, wb_g1, nullptr, h1, nullptr, U_N, 512);
  gemm(ux_bf, wb_g2, nullptr, h2, nullptr, U_N, 512);

  // 4. fused GRU + self-excitation dots (WMMA)
  gru_kernel<<<dim3(NS_N / 16), dim3(256), 0, stream>>>(
      ux_bf, u_series, wb_ih, wb_hh, b_ih, b_hh, w_his, w_intra, wall, fband);

  // 5. pofe
  fill(gi_s, 0.f, 1);
  fill(pofe, 0.f, M_N);
  mdots_kernel<<<dim3(M_N), dim3(128), 0, stream>>>(mx_f, w_base, w_inter, mft, mdb,
                                                    mdi, gi_s);
  pofe_kernel<<<dim3(NS_N / 256), dim3(256), 0, stream>>>(wall, fband, ts, slt, sli,
                                                          seq_idx, mdb, mdi, gi_s, pofe);
  pofe_norm_kernel<<<dim3(1), dim3(512), 0, stream>>>(pofe, pofen);

  // 6. attention vectors for both sparse-softmax layers
  attvec_kernel<<<dim3((NN_N + 255) / 256), dim3(256), 0, stream>>>(
      px_f, poc_att, int_att, asrc, adst, bsrc, bdst, NN_N);

  // 7. pof_learning (src >= U)
  fill(rmax, -HUGE_VALF, M_N);
  fill(rsum, 0.f, M_N);
  fill(pm, 0.f, M_N * 128);
  pof_edge_vals_kernel<<<dim3((EPOF + 255) / 256), dim3(256), 0, stream>>>(
      pof_ei, asrc, adst, rt, t_o, ev, rmax);
  pof_edge_exp_kernel<<<dim3((EPOF + 255) / 256), dim3(256), 0, stream>>>(pof_ei, ev,
                                                                          rmax, ee, rsum);
  pof_edge_acc_kernel<<<dim3(EPOF), dim3(128), 0, stream>>>(pof_ei, ee, rsum, px_f, pm);

  // 8. user_interest_conv (src < U; self loops in init)
  int_self_init_kernel<<<dim3((U_N + 255) / 256), dim3(256), 0, stream>>>(bsrc, bdst,
                                                                          selfv, imax);
  int_edge_max_kernel<<<dim3((EPOF + 255) / 256), dim3(256), 0, stream>>>(pof_ei, bsrc,
                                                                          bdst, imax);
  int_sum_init_kernel<<<dim3((U_N + 255) / 256), dim3(256), 0, stream>>>(selfv, imax,
                                                                         isum);
  int_edge_sum_kernel<<<dim3((EPOF + 255) / 256), dim3(256), 0, stream>>>(pof_ei, bsrc,
                                                                          bdst, imax, isum);
  fill(uacc, 0.f, U_N * 128);
  int_edge_acc_kernel<<<dim3(EPOF), dim3(128), 0, stream>>>(pof_ei, bsrc, bdst, imax,
                                                            isum, px_f, pm, pofen, uacc);
  int_self_acc_kernel<<<dim3(U_N), dim3(128), 0, stream>>>(selfv, imax, isum, px_f, uacc);

  // 9. GAT 1 (pr edges) and GAT 2 (friend edges)
  const int* gat_ei[2] = {pr_ei, fr_ei};
  float* gs[2] = {s1, s2};
  float* gd[2] = {d1, d2};
  float* gm[2] = {dmax1, dmax2};
  float* gu[2] = {dsum1, dsum2};
  float* gh[2] = {h1, h2};
  float* ga[2] = {gacc1, gacc2};
  const float* gaspar[2] = {g1as, g2as};
  const float* gadpar[2] = {g1ad, g2ad};
  for (int g = 0; g < 2; ++g) {
    gat_sd_kernel<<<dim3((U_N * 4 + 255) / 256), dim3(256), 0, stream>>>(
        gh[g], gaspar[g], gadpar[g], gs[g], gd[g], U_N);
    gat_self_init_kernel<<<dim3((U_N * 4 + 255) / 256), dim3(256), 0, stream>>>(
        gs[g], gd[g], gm[g], U_N);
    gat_edge_max_kernel<<<dim3((EGAT + 255) / 256), dim3(256), 0, stream>>>(
        gat_ei[g], EGAT, gs[g], gd[g], gm[g]);
    gat_sum_init_kernel<<<dim3((U_N * 4 + 255) / 256), dim3(256), 0, stream>>>(
        gs[g], gd[g], gm[g], gu[g], U_N);
    gat_edge_sum_kernel<<<dim3((EGAT + 255) / 256), dim3(256), 0, stream>>>(
        gat_ei[g], EGAT, gs[g], gd[g], gm[g], gu[g]);
    fill(ga[g], 0.f, U_N * 128);
    gat_edge_acc_kernel<<<dim3(EGAT), dim3(128), 0, stream>>>(gat_ei[g], EGAT, gs[g],
                                                              gd[g], gm[g], gu[g], gh[g],
                                                              ga[g]);
    gat_self_acc_kernel<<<dim3(U_N), dim3(128), 0, stream>>>(gs[g], gd[g], gm[g], gu[g],
                                                             gh[g], ga[g]);
  }

  // 10. fused single-query attention (WMMA) -> long_msg accumulation
  fill(longacc, 0.f, M_N * 128);
  attn_kernel<<<dim3(NS_N), dim3(256), 0, stream>>>(ux_bf, mx_f, u_series, seq_idx, ts,
                                                    mask, wb_K, wb_V, QW, Qb, Kb, Vb,
                                                    ftW, ftb, longacc);

  // 11. outputs
  float* out = (float*)d_out;
  out_msg_kernel<<<dim3((M_N * 128 + 255) / 256), dim3(256), 0, stream>>>(pm, longacc,
                                                                          out);
  out_user_kernel<<<dim3((U_N * 128 + 255) / 256), dim3(256), 0, stream>>>(
      gacc1, g1b, uacc, gacc2, g2b, out + (size_t)M_N * 128);
}
